// CifMiddleware_54735063220513
// MI455X (gfx1250) — matrix operand order repaired and checked by reading, not compiled, not run on hardware
//
#include <hip/hip_runtime.h>
#include <hip/hip_bf16.h>
#include <math.h>

typedef __attribute__((ext_vector_type(16))) _Float16 v16h;
typedef __attribute__((ext_vector_type(8)))  float    v8f;

#define BB 16
#define TT 4096
#define CC 256

// pack 8 contiguous floats (16B-aligned) into v16h elements [i0, i0+8)
__device__ __forceinline__ void pack8(v16h& v, int i0, const float* p) {
    float4 a = *(const float4*)p;
    float4 b = *(const float4*)(p + 4);
    v[i0 + 0] = (_Float16)a.x; v[i0 + 1] = (_Float16)a.y;
    v[i0 + 2] = (_Float16)a.z; v[i0 + 3] = (_Float16)a.w;
    v[i0 + 4] = (_Float16)b.x; v[i0 + 5] = (_Float16)b.y;
    v[i0 + 6] = (_Float16)b.z; v[i0 + 7] = (_Float16)b.w;
}

// ---------------------------------------------------------------------------
// Kernel 0: pre-pack dense_W (f32, row-major) into per-lane f16 B-fragment
// layout: Wh[((nt*8 + kb)*32 + lane)*16 + i] = W[d][k],
//   d = nt*16 + (lane&15),  k = kb*32 + (lane>>4)*16 + i      (128 KB total)
// ---------------------------------------------------------------------------
__global__ __launch_bounds__(256) void cif_pack_W(
    const float* __restrict__ dense_W,   // (C,C)
    _Float16* __restrict__ Wh)           // 16*8*32*16 halves
{
    const int idx  = blockIdx.x * 256 + threadIdx.x;   // 0 .. 65535
    const int i    = idx & 15;
    const int lane = (idx >> 4) & 31;
    const int kb   = (idx >> 9) & 7;
    const int nt   = idx >> 12;
    const int d = nt * 16 + (lane & 15);
    const int k = kb * 32 + (lane >> 4) * 16 + i;
    Wh[idx] = (_Float16)dense_W[(size_t)d * CC + k];
}

// ---------------------------------------------------------------------------
// Kernel 1: org_w[b,t] = sigmoid( wproj . relu(x W^T + bias) ) * not_pad
// One wave per 16-timestep M-tile, 4 waves/block. The packed B table (128 KB)
// is staged once per workgroup into LDS (320 KB WGP LDS) and all WMMAs feed
// from ds_load_b128; A fragments hoisted in VGPRs (K=256, f32->f16 once).
// ---------------------------------------------------------------------------
__global__ __launch_bounds__(128) void cif_weight_wmma(
    const float*    __restrict__ x,        // (B,T,C)
    const _Float16* __restrict__ Wh,       // packed B frags (global)
    const float*    __restrict__ dense_b,  // (C)
    const float*    __restrict__ wproj_W,  // (C)
    const float*    __restrict__ wproj_b,  // (1)
    const int*      __restrict__ lengths,  // (B)
    float* __restrict__ org_w)             // (B,T)
{
    __shared__ __align__(16) _Float16 shW[16 * 8 * 32 * 16];   // 128 KB

    const int tid   = threadIdx.x;
    const int lane  = tid & 31;
    const int wave  = tid >> 5;
    const int mtile = blockIdx.x * 4 + wave;   // 0 .. B*T/16-1
    const int b  = mtile >> 8;                 // T/16 = 256 tiles per batch
    const int t0 = (mtile & 255) << 4;
    const int nlo = lane & 15;                 // column / row-in-half index
    const int hi  = lane >> 4;                 // which 16-lane half

    // ---- cooperative stage of packed B into LDS (8192 x 16B) ----
    {
        uint4* dst = (uint4*)shW;
        const uint4* src = (const uint4*)Wh;
        for (int i = tid; i < 8192; i += 128)
            dst[i] = src[i];
    }

    // ---- load all A fragments: rows t0..t0+15, K = 0..255 (f32 -> f16) ----
    const float* xm = x + ((size_t)b * TT + (t0 + nlo)) * CC;
    v16h A[8];
    #pragma unroll
    for (int kb = 0; kb < 8; ++kb) {
        const float* p = xm + kb * 32 + hi * 16;
        pack8(A[kb], 0, p);        // K = kb*32 + hi*16 + [0..7]
        pack8(A[kb], 8, p + 8);    // K = kb*32 + hi*16 + 8 + [0..7]
    }

    __syncthreads();
    const v16h* Bfrag = (const v16h*)shW;      // 32B-aligned LDS fragments

    // ---- N loop: 16 tiles of 16 output channels each ----
    float s[8];
    #pragma unroll
    for (int r = 0; r < 8; ++r) s[r] = 0.0f;

    for (int nt = 0; nt < 16; ++nt) {
        const v16h* Bp = Bfrag + (nt * 8) * 32 + lane;
        v8f acc0 = {}, acc1 = {};
        #pragma unroll
        for (int kb = 0; kb < 8; kb += 2) {
            v16h B0 = Bp[kb * 32];
            v16h B1 = Bp[(kb + 1) * 32];
            acc0 = __builtin_amdgcn_wmma_f32_16x16x32_f16(
                false, A[kb], false, B0, (short)0, acc0, false, false);
            acc1 = __builtin_amdgcn_wmma_f32_16x16x32_f16(
                false, A[kb + 1], false, B1, (short)0, acc1, false, false);
        }
        const int d = nt * 16 + nlo;
        const float bias = dense_b[d];
        const float wp   = wproj_W[d];
        #pragma unroll
        for (int r = 0; r < 8; ++r) {
            float h = (acc0[r] + acc1[r]) + bias;
            h = fmaxf(h, 0.0f);
            s[r] = fmaf(h, wp, s[r]);
        }
    }

    // ---- reduce over the 16 lanes of each half (N dimension) ----
    #pragma unroll
    for (int off = 1; off <= 8; off <<= 1) {
        #pragma unroll
        for (int r = 0; r < 8; ++r)
            s[r] += __shfl_xor(s[r], off, 32);
    }

    // lane half 'hi', element r holds sum for m = hi*8 + r
    const float wpb = wproj_b[0];
    const int len = lengths[b];
    #pragma unroll
    for (int r = 0; r < 8; ++r) {
        if (nlo == r) {
            const int t = t0 + hi * 8 + r;
            float sv = s[r] + wpb;
            float wv = 1.0f / (1.0f + expf(-sv));
            if (t >= len) wv = 0.0f;               // not_pad mask
            org_w[(size_t)b * TT + t] = wv;
        }
    }
}

// ---------------------------------------------------------------------------
// Kernel 2: per-batch scalar scan in LDS.
// Produces: quantity_out, res_w, fire_mask, cif_mask (direct to d_out) and
// coefA / coefR / packed(fired|pos) + nfired into workspace.
// ---------------------------------------------------------------------------
__global__ __launch_bounds__(256) void cif_scalar_scan(
    const float* __restrict__ org_w,    // (B,T)
    const int*   __restrict__ lengths,  // (B)
    const float* __restrict__ tlen,     // (B)
    float* __restrict__ coefA,          // (B,T) ws
    float* __restrict__ coefR,          // (B,T) ws
    int*   __restrict__ mpk,            // (B,T) ws: bit30=fired, low=pos or 0x3FFFFFFF
    int*   __restrict__ nfired,         // (B)   ws
    float* __restrict__ out_quant,      // (B)
    float* __restrict__ out_resw,       // (B)
    float* __restrict__ out_cifmask,    // (B,T)
    float* __restrict__ out_firemask)   // (B,T)
{
    __shared__ float shw[TT];
    __shared__ float shr[TT];
    __shared__ int   shm[TT];
    __shared__ float ssum[256];
    __shared__ int   snf;

    const int b = blockIdx.x;
    const int tid = threadIdx.x;

    float lsum = 0.0f;
    for (int t = tid; t < TT; t += 256) {
        float v = org_w[(size_t)b * TT + t];
        shw[t] = v;
        lsum += v;
    }
    ssum[tid] = lsum;
    __syncthreads();
    for (int st = 128; st > 0; st >>= 1) {
        if (tid < st) ssum[tid] += ssum[tid + st];
        __syncthreads();
    }

    if (tid == 0) {
        const float S = ssum[0];
        out_quant[b] = S;                              // quantity_out
        const float scale = tlen[b] / S;
        int ps = lengths[b]; if (ps > TT) ps = TT;     // pad_start
        int last = ps - 1; if (last < 0) last = 0;

        float acc = 0.0f, resw = 0.0f;
        int cnt = 0;
        for (int t = 0; t < TT; ++t) {
            const float w = shw[t] * scale;
            const bool fired = (acc + w) >= 1.0f;
            const float rem = 1.0f - acc;
            float cA;
            if (fired) { cA = w - rem; acc = cA; }
            else       { cA = w;       acc = acc + w; }
            const bool mk = fired && (t <= ps);        // i > pad_start -> zeroed
            int enc = fired ? (1 << 30) : 0;
            enc |= mk ? cnt : 0x3FFFFFFF;
            if (mk) cnt++;
            shw[t] = cA;
            shr[t] = rem;
            shm[t] = enc;
            if (t == last) resw = acc;
        }
        out_resw[b] = resw;
        nfired[b] = cnt;
        snf = cnt;
    }
    __syncthreads();

    const int nf = snf;
    for (int t = tid; t < TT; t += 256) {
        const size_t o = (size_t)b * TT + t;
        coefA[o] = shw[t];
        coefR[o] = shr[t];
        const int enc = shm[t];
        mpk[o] = enc;
        out_firemask[o] = (enc & (1 << 30)) ? 1.0f : 0.0f;
        out_cifmask[o]  = (t < nf) ? 1.0f : 0.0f;
    }
}

// ---------------------------------------------------------------------------
// Kernel 3: channel-parallel acc_h scan + compacted scatter of fired frames.
// 8 blocks x 32 lanes per batch; thread owns one channel. Prefetch hides the
// HBM latency of the dependent serial loop.
// ---------------------------------------------------------------------------
__global__ __launch_bounds__(32) void cif_channel_scan(
    const float* __restrict__ x,       // (B,T,C)
    const float* __restrict__ coefA,
    const float* __restrict__ coefR,
    const int*   __restrict__ mpk,
    const int*   __restrict__ nfired,
    const int*   __restrict__ lengths,
    float* __restrict__ cif_out,       // (B,T,C)
    float* __restrict__ res_h)         // (B,C)
{
    const int b  = blockIdx.x >> 3;
    const int cb = blockIdx.x & 7;
    const int c  = cb * 32 + threadIdx.x;

    const float* xb = x + (size_t)b * TT * CC + c;
    const float* cA = coefA + (size_t)b * TT;
    const float* cR = coefR + (size_t)b * TT;
    const int*   mp = mpk   + (size_t)b * TT;
    float* cob = cif_out + (size_t)b * TT * CC + c;

    int ps = lengths[b]; if (ps > TT) ps = TT;
    int last = ps - 1; if (last < 0) last = 0;

    float acc = 0.0f, rh = 0.0f;
    for (int t = 0; t < TT; ++t) {
        if (t + 32 < TT)
            __builtin_prefetch(xb + (size_t)(t + 32) * CC, 0, 0);
        const float e = xb[(size_t)t * CC];
        const float a = cA[t];
        const float r = cR[t];
        const int   m = mp[t];
        if (m & (1 << 30)) {                       // fired
            const float f = fmaf(r, e, acc);
            const int pos = m & 0x3FFFFFFF;
            if (pos != 0x3FFFFFFF)
                cob[(size_t)pos * CC] = f;         // compacted scatter
            acc = a * e;                           // reset: (w - remained) * e
        } else {
            acc = fmaf(a, e, acc);                 // accumulate: + w * e
        }
        if (t == last) rh = acc;
    }
    res_h[(size_t)b * CC + c] = rh;

    // zero-fill columns past the fired count
    const int nf = nfired[b];
    for (int j = nf; j < TT; ++j)
        cob[(size_t)j * CC] = 0.0f;
}

// ---------------------------------------------------------------------------
extern "C" void kernel_launch(void* const* d_in, const int* in_sizes, int n_in,
                              void* d_out, int out_size, void* d_ws, size_t ws_size,
                              hipStream_t stream) {
    const float* x        = (const float*)d_in[0];
    const int*   lengths  = (const int*)  d_in[1];
    const float* tlen     = (const float*)d_in[2];
    const float* dense_W  = (const float*)d_in[3];
    const float* dense_b  = (const float*)d_in[4];
    const float* wproj_W  = (const float*)d_in[5];
    const float* wproj_b  = (const float*)d_in[6];

    float* out = (float*)d_out;
    const size_t OFF_CMASK = (size_t)BB * TT * CC;
    const size_t OFF_QUANT = OFF_CMASK + (size_t)BB * TT;
    const size_t OFF_RESW  = OFF_QUANT + BB;
    const size_t OFF_RESH  = OFF_RESW + BB;
    const size_t OFF_FMASK = OFF_RESH + (size_t)BB * CC;

    // workspace layout: Wh (128KB, aligned at base) | org_w | coefA | coefR | mpk | nfired
    _Float16* Wh = (_Float16*)d_ws;                     // 16*8*32*16 = 65536 halves
    float* org_w = (float*)((char*)d_ws + 65536 * sizeof(_Float16));
    float* coefA = org_w + (size_t)BB * TT;
    float* coefR = coefA + (size_t)BB * TT;
    int*   mpk   = (int*)(coefR + (size_t)BB * TT);
    int*   nfire = mpk + (size_t)BB * TT;

    // 0) pack dense_W into f16 fragment layout (128KB, L2-resident)
    cif_pack_W<<<256, 256, 0, stream>>>(dense_W, Wh);

    // 1) WMMA weight producer: 4096 M-tiles, 4 waves/block, B staged in LDS
    cif_weight_wmma<<<(BB * TT / 16) / 4, 128, 0, stream>>>(
        x, Wh, dense_b, wproj_W, wproj_b, lengths, org_w);

    // 2) per-batch scalar scan (LDS-resident)
    cif_scalar_scan<<<BB, 256, 0, stream>>>(
        org_w, lengths, tlen, coefA, coefR, mpk, nfire,
        out + OFF_QUANT, out + OFF_RESW, out + OFF_CMASK, out + OFF_FMASK);

    // 3) channel-parallel vector scan + compaction
    cif_channel_scan<<<BB * 8, 32, 0, stream>>>(
        x, coefA, coefR, mpk, nfire, lengths,
        out /*OFF_CIF=0*/, out + OFF_RESH);
}
